// DynamicMaskHead_46351287059000
// MI455X (gfx1250) — matrix operand order repaired and checked by reading, not compiled, not run on hardware
//
#include <hip/hip_runtime.h>
#include <hip/hip_bf16.h>
#include <math.h>

typedef __attribute__((ext_vector_type(16))) _Float16 v16h;
typedef __attribute__((ext_vector_type(8)))  float    v8f;

#define HH    100
#define WWID  152
#define HWSZ  (HH*WWID)      // 15200 = 950*16, tiles divide exactly
#define NPARAM 187
#define OH    200
#define OW    304

// ---------------------------------------------------------------------------
// Kernel 1: per-instance 3-layer MLP over all pixels via WMMA f16 -> f32.
//   M = out channels (pad 16), K = in channels (pad 32), N = 16 pixels/tile.
//   grid = (n_inst, SPLIT_Y), 256 threads = 8 waves; each wave owns a strided
//   set of 16-pixel tiles. Weights/biases live in registers for the loop.
// ---------------------------------------------------------------------------
__global__ __launch_bounds__(256) void dmh_mlp_wmma(
    const float* __restrict__ mask_feats,   // (4, 8, H, W)
    const float* __restrict__ params,       // (n_inst, 187)
    const float* __restrict__ ilocs,        // (n_inst, 2)
    const float* __restrict__ soi,          // (5,)
    const int*   __restrict__ im_inds,      // (n_inst,)
    const int*   __restrict__ fpn_levels,   // (n_inst,)
    const int*   __restrict__ stride_p,     // scalar
    float*       __restrict__ logits)       // (n_inst, 3, H*W) scratch
{
  const int inst = blockIdx.x;
  const int lane = threadIdx.x & 31;
  const int wave = threadIdx.x >> 5;       // 0..7
  const int m    = lane & 15;              // row within A / col within B,D
  const int grp  = lane >> 4;              // lane-half

  const float* p = params + (size_t)inst * NPARAM;

  // --- A matrices (weights), 16-bit A layout: lane m, element i -> K ---
  v16h a1, a2, a3;
  #pragma unroll
  for (int i = 0; i < 16; ++i) {
    const int j  = i & 7;
    const int hf = i >> 3;
    const int k  = i + 16 * grp;                          // contiguous K map (matches B)
    const float w1 = (m < 8 && k < 10) ? p[       m * 10 + k] : 0.0f;
    const float w2 = (m < 8 && k <  8) ? p[ 80 +  m *  8 + k] : 0.0f;
    const float w3 = (m < 3 && k <  8) ? p[144 +  m *  8 + k] : 0.0f;
    a1[i] = (_Float16)w1;
    a2[i] = (_Float16)w2;
    a3[i] = (_Float16)w3;
  }

  // --- C matrices (biases): VGPR r -> M = r + 8*grp; pad rows get 0 ---
  v8f c1, c2, c3;
  #pragma unroll
  for (int r = 0; r < 8; ++r) {
    c1[r] = (grp == 0) ? p[168 + r] : 0.0f;
    c2[r] = (grp == 0) ? p[176 + r] : 0.0f;
    c3[r] = (grp == 0 && r < 3) ? p[184 + r] : 0.0f;
  }

  const int   im      = im_inds[inst];
  const float inv_soi = 1.0f / soi[fpn_levels[inst]];
  const float locx    = ilocs[2 * inst + 0];
  const float locy    = ilocs[2 * inst + 1];
  const int   stride  = *stride_p;
  const float shalf   = (float)(stride >> 1);

  const float* fb   = mask_feats + (size_t)im * 8 * HWSZ;
  float*       lout = logits     + (size_t)inst * 3 * HWSZ;

  const int ntiles = HWSZ / 16;   // 950
  for (int tile = wave + blockIdx.y * 8; tile < ntiles; tile += 8 * gridDim.y) {
    const int p0  = tile * 16;
    const int pix = p0 + m;                  // pixel for column n = m
    const int py  = pix / WWID;
    const int px  = pix - py * WWID;

    // B (32x16 f16): lanes 0-15 carry K=0..15 (10 real channels),
    // lanes 16-31 carry K=16..31 == 0 padding.
    v16h b = {};
    if (grp == 0) {
      b[0] = (_Float16)((locx - ((float)(px * stride) + shalf)) * inv_soi);
      b[1] = (_Float16)((locy - ((float)(py * stride) + shalf)) * inv_soi);
      #pragma unroll
      for (int c = 0; c < 8; ++c)
        b[2 + c] = (_Float16)fb[c * HWSZ + pix];
    }

    v8f d1 = __builtin_amdgcn_wmma_f32_16x16x32_f16(
        false, a1, false, b, (short)0, c1, false, false);

    // relu + repack as next B. For grp1 lanes d1[] holds padded rows == 0,
    // and their needed K=16..31 are 0, so this is uniform across lanes.
    v16h b2;
    #pragma unroll
    for (int i = 0; i < 16; ++i)
      b2[i] = (i < 8) ? (_Float16)fmaxf(d1[i], 0.0f) : (_Float16)0.0f;

    v8f d2 = __builtin_amdgcn_wmma_f32_16x16x32_f16(
        false, a2, false, b2, (short)0, c2, false, false);

    v16h b3;
    #pragma unroll
    for (int i = 0; i < 16; ++i)
      b3[i] = (i < 8) ? (_Float16)fmaxf(d2[i], 0.0f) : (_Float16)0.0f;

    v8f d3 = __builtin_amdgcn_wmma_f32_16x16x32_f16(
        false, a3, false, b3, (short)0, c3, false, false);

    // D: VGPR r, lanes 0-15 -> M=r. Output channels 0..2 live in grp0.
    if (grp == 0) {
      float* lp = lout + p0 + m;
      lp[0]        = d3[0];
      lp[HWSZ]     = d3[1];
      lp[2 * HWSZ] = d3[2];
    }
  }
}

// ---------------------------------------------------------------------------
// Kernel 2: aligned bilinear x2 (closed form) + channel split + sigmoid.
// One thread per (inst, oy, ox); coalesced stores; logits stay in L2.
// ---------------------------------------------------------------------------
__global__ __launch_bounds__(256) void dmh_upsample(
    const float* __restrict__ logits, float* __restrict__ out, int n_inst)
{
  const int idx   = blockIdx.x * 256 + threadIdx.x;
  const int total = n_inst * OH * OW;
  if (idx >= total) return;

  const int ox   = idx % OW;
  int t          = idx / OW;
  const int oy   = t % OH;
  const int inst = t / OH;

  // out[:, :, max(arange(OH)-1,0)] sampling at position yi/2, edge-padded.
  const int   yi = (oy > 1) ? (oy - 1) : 0;
  const int   xi = (ox > 1) ? (ox - 1) : 0;
  const int   y0 = yi >> 1;
  const int   x0 = xi >> 1;
  const float fy = (yi & 1) ? 0.5f : 0.0f;
  const float fx = (xi & 1) ? 0.5f : 0.0f;
  const int   y1 = (y0 + 1 < HH)   ? (y0 + 1) : (HH - 1);
  const int   x1 = (x0 + 1 < WWID) ? (x0 + 1) : (WWID - 1);

  const float w00 = (1.0f - fy) * (1.0f - fx);
  const float w01 = (1.0f - fy) * fx;
  const float w10 = fy * (1.0f - fx);
  const float w11 = fy * fx;

  const float* L = logits + (size_t)inst * 3 * HWSZ;
  const size_t out1_base = (size_t)n_inst * 2 * OH * OW;

  #pragma unroll
  for (int c = 0; c < 3; ++c) {
    const float* Lc = L + c * HWSZ;
    const float v = w00 * Lc[y0 * WWID + x0] + w01 * Lc[y0 * WWID + x1]
                  + w10 * Lc[y1 * WWID + x0] + w11 * Lc[y1 * WWID + x1];
    if (c < 2) {
      out[(((size_t)inst * 2 + c) * OH + oy) * OW + ox] = v;
    } else {
      out[out1_base + ((size_t)inst * OH + oy) * OW + ox] =
          1.0f / (1.0f + expf(-v));
    }
  }
}

extern "C" void kernel_launch(void* const* d_in, const int* in_sizes, int n_in,
                              void* d_out, int out_size, void* d_ws, size_t ws_size,
                              hipStream_t stream) {
  const float* mask_feats = (const float*)d_in[0];
  const float* params     = (const float*)d_in[1];
  const float* ilocs      = (const float*)d_in[2];
  const float* soi        = (const float*)d_in[3];
  const int*   im_inds    = (const int*)  d_in[4];
  const int*   fpn        = (const int*)  d_in[5];
  const int*   stride_p   = (const int*)  d_in[6];

  float* out    = (float*)d_out;
  float* logits = (float*)d_ws;            // n_inst*3*HWSZ floats (~36.5 MB)

  const int n_inst = in_sizes[4];          // im_inds count

  // 5-way split in Y for more waves in flight; weight reload cost is trivial.
  dim3 grid1(n_inst, 5);
  dmh_mlp_wmma<<<grid1, 256, 0, stream>>>(mask_feats, params, ilocs, soi,
                                          im_inds, fpn, stride_p, logits);

  const int total = n_inst * OH * OW;
  dmh_upsample<<<(total + 255) / 256, 256, 0, stream>>>(logits, out, n_inst);
}
